// QNN_73005854097728
// MI455X (gfx1250) — matrix operand order repaired and checked by reading, not compiled, not run on hardware
//
#include <hip/hip_runtime.h>
#include <hip/hip_bf16.h>

typedef __attribute__((ext_vector_type(16))) _Float16 v16h;
typedef __attribute__((ext_vector_type(8)))  float    v8f;
typedef __attribute__((__vector_size__(4 * sizeof(int)))) int v4i;

#define NQ    10
#define DIM   1024
#define BATCH 8192
#define MDIM  2048   // 2*DIM real block rows
#define KDIM  2048   // 2*DIM real block cols

// Static device scratch (graph-capture safe; fully rewritten every call).
__device__ __align__(16) _Float16 g_W[(size_t)MDIM * KDIM];     // 8 MB  [[ReU,-ImU],[ImU,ReU]]
__device__ __align__(16) _Float16 g_PsiT[(size_t)BATCH * KDIM]; // 32 MB  circuit-major [c][k]

#if __has_builtin(__builtin_amdgcn_global_load_async_to_lds_b128)
#define HAS_ASYNC_LDS 1
#else
#define HAS_ASYNC_LDS 0
#endif

// Pointer casts matching the builtin prototype revealed by the round-2 diagnostic:
//   param0: int4 addrspace(1)*  (non-const)   param1: addrspace(3) pointer
#define GPTR_V4I(p) ((__attribute__((address_space(1))) v4i*)(p))
#define LPTR_V4I(p) ((__attribute__((address_space(3))) v4i*)(p))

__device__ inline void wait_async0(){
#if __has_builtin(__builtin_amdgcn_s_wait_asynccnt)
  __builtin_amdgcn_s_wait_asynccnt(0);
#else
  asm volatile("s_wait_asynccnt 0" ::: "memory");
#endif
}

// ---------------- complex helpers ----------------
__device__ inline float2 cmul(float2 a, float2 b){
  return make_float2(a.x*b.x - a.y*b.y, a.x*b.y + a.y*b.x);
}

template<int MASK>
__device__ inline float2 swz2(float2 v){
  // group-of-32 ds_swizzle: and=0x1f, or=0, xor=MASK  -> lane reads lane^MASK
  constexpr int OFF = (MASK << 10) | 0x1f;
  float2 r;
  r.x = __int_as_float(__builtin_amdgcn_ds_swizzle(__float_as_int(v.x), OFF));
  r.y = __int_as_float(__builtin_amdgcn_ds_swizzle(__float_as_int(v.y), OFF));
  return r;
}

// Single-qubit gate on qubit J (qubit0 = MSB of state index).
// State index s = (lane<<5) | r  ->  qubits 0..4 on lane bits 4..0, qubits 5..9 on reg bits 4..0.
template<int J>
__device__ inline void apply1(float2 (&a)[32], int lane,
                              float2 m00, float2 m01, float2 m10, float2 m11){
  if constexpr (J <= 4){
    constexpr int BL   = 4 - J;
    constexpr int MASK = 1 << BL;
    const int bit = (lane >> BL) & 1;
    const float2 cs = make_float2(bit ? m11.x : m00.x, bit ? m11.y : m00.y);
    const float2 co = make_float2(bit ? m10.x : m01.x, bit ? m10.y : m01.y);
#pragma unroll
    for (int r = 0; r < 32; ++r){
      float2 b  = swz2<MASK>(a[r]);
      float2 t0 = cmul(cs, a[r]);
      float2 t1 = cmul(co, b);
      a[r] = make_float2(t0.x + t1.x, t0.y + t1.y);
    }
  } else {
    constexpr int MASK = 1 << (9 - J);
#pragma unroll
    for (int r = 0; r < 32; ++r){
      if ((r & MASK) == 0){
        const int r1 = r | MASK;
        float2 a0 = a[r], a1 = a[r1];
        float2 u0 = cmul(m00, a0), u1 = cmul(m01, a1);
        float2 v0 = cmul(m10, a0), v1 = cmul(m11, a1);
        a[r]  = make_float2(u0.x + u1.x, u0.y + u1.y);
        a[r1] = make_float2(v0.x + v1.x, v0.y + v1.y);
      }
    }
  }
}

__device__ inline float2 xxmix(float2 a, float2 b, float c, float s){
  // c*a - i*s*b
  return make_float2(c*a.x + s*b.y, c*a.y - s*b.x);
}

// RXX on qubits (J, J+1): partner = flip both bits.
template<int J>
__device__ inline void apply_rxx(float2 (&a)[32], float c, float s){
  if constexpr (J <= 3){
    constexpr int MASK = (1 << (4 - J)) | (1 << (3 - J));       // both bits on lane axis
#pragma unroll
    for (int r = 0; r < 32; ++r){
      float2 b = swz2<MASK>(a[r]);
      a[r] = xxmix(a[r], b, c, s);
    }
  } else if constexpr (J == 4){                                  // straddle lane bit0 / reg bit4
#pragma unroll
    for (int r = 0; r < 16; ++r){
      const int r1 = r + 16;
      float2 b0 = swz2<1>(a[r1]);
      float2 b1 = swz2<1>(a[r]);
      a[r]  = xxmix(a[r],  b0, c, s);
      a[r1] = xxmix(a[r1], b1, c, s);
    }
  } else {                                                       // both bits on reg axis
    constexpr int H    = 1 << (9 - J);
    constexpr int MASK = H | (1 << (8 - J));
#pragma unroll
    for (int r = 0; r < 32; ++r){
      if ((r & H) == 0){
        const int r1 = r ^ MASK;
        float2 t0 = a[r], t1 = a[r1];
        a[r]  = xxmix(t0, t1, c, s);
        a[r1] = xxmix(t1, t0, c, s);
      }
    }
  }
}

// tensorcircuit r-gate: exp(-i t n.sigma), n=(sin a cos p, sin a sin p, cos a)
__device__ inline void rmat(float t, float p, float al,
                            float2& m00, float2& m01, float2& m10, float2& m11){
  float ct = cosf(t),  st = sinf(t);
  float ca = cosf(al), sa = sinf(al);
  float cp = cosf(p),  sp = sinf(p);
  float ss = st * sa;
  m00 = make_float2(ct, -st * ca);
  m01 = make_float2(-ss * sp, -ss * cp);   // -i*st*sa*e^{-ip}
  m10 = make_float2( ss * sp, -ss * cp);   // -i*st*sa*e^{+ip}
  m11 = make_float2(ct,  st * ca);
}

// ---------------- phase 0: out[c] = bias ----------------
__global__ void qnn_init_out(const float* __restrict__ bias, float* __restrict__ out){
  int i = blockIdx.x * blockDim.x + threadIdx.x;
  if (i < BATCH) out[i] = bias[0];
}

// ---------------- phase 1: build W (x-independent circuit as a matrix) ----------------
// 1024 waves; wave t simulates U|t> with the per-wave register state-vector engine.
__global__ __launch_bounds__(256) void qnn_build_W(const float* __restrict__ qw){
  const int lane = threadIdx.x & 31;
  const int t    = (blockIdx.x * blockDim.x + threadIdx.x) >> 5;   // basis column 0..1023
  const int tl = t >> 5, tr = t & 31;
  float2 a[32];
#pragma unroll
  for (int r = 0; r < 32; ++r)
    a[r] = make_float2((lane == tl && r == tr) ? 1.0f : 0.0f, 0.0f);

  float2 m00, m01, m10, m11;
#define RGATE(J, BASE)                                                         \
  rmat(qw[(BASE) + (J)*3 + 0], qw[(BASE) + (J)*3 + 1], qw[(BASE) + (J)*3 + 2], \
       m00, m01, m10, m11);                                                    \
  apply1<J>(a, lane, m00, m01, m10, m11);
#define XXGATE(J) { float th = qw[30 + (J)]; apply_rxx<J>(a, cosf(th), sinf(th)); }

  // R layer (weights), offset 0
  RGATE(0,0) RGATE(1,0) RGATE(2,0) RGATE(3,0) RGATE(4,0)
  RGATE(5,0) RGATE(6,0) RGATE(7,0) RGATE(8,0) RGATE(9,0)
  // RXX chain (ent_w), offset 30  (all XX terms commute; sequential order matches ref)
  XXGATE(0) XXGATE(1) XXGATE(2) XXGATE(3) XXGATE(4)
  XXGATE(5) XXGATE(6) XXGATE(7) XXGATE(8)
  // final R layer, offset 39
  RGATE(0,39) RGATE(1,39) RGATE(2,39) RGATE(3,39) RGATE(4,39)
  RGATE(5,39) RGATE(6,39) RGATE(7,39) RGATE(8,39) RGATE(9,39)
#undef RGATE
#undef XXGATE

  // Emit real block matrix column t (and t+DIM)
#pragma unroll
  for (int r = 0; r < 32; ++r){
    const int s = (lane << 5) | r;
    _Float16 re = (_Float16)a[r].x;
    _Float16 im = (_Float16)a[r].y;
    g_W[(size_t)s * KDIM + t]                 = re;
    g_W[(size_t)(s + DIM) * KDIM + t]         = im;
    g_W[(size_t)s * KDIM + (t + DIM)]         = (_Float16)(-a[r].y);
    g_W[(size_t)(s + DIM) * KDIM + (t + DIM)] = re;
  }
}

// ---------------- phase 2: product states Psi0 (circuit-major, [c][2048]) ----------------
__global__ __launch_bounds__(256) void qnn_build_psi(const float* __restrict__ x,
                                                     const float* __restrict__ fm){
  const int c = blockIdx.x;
  __shared__ float cs[NQ], sn[NQ];
  if (threadIdx.x < NQ){
    float th = 0.5f * x[(size_t)c * NQ + threadIdx.x] * fm[0];
    cs[threadIdx.x] = cosf(th);
    sn[threadIdx.x] = sinf(th);
  }
  __syncthreads();
  for (int s = threadIdx.x; s < DIM; s += blockDim.x){
    float m = 1.0f;
#pragma unroll
    for (int j = 0; j < NQ; ++j){
      m *= ((s >> (9 - j)) & 1) ? sn[j] : cs[j];
    }
    const int p = __popc(s) & 3;               // phase (-i)^p
    float re = (p == 0) ? m : (p == 2 ? -m : 0.0f);
    float im = (p == 1) ? -m : (p == 3 ?  m : 0.0f);
    g_PsiT[(size_t)c * KDIM + s]       = (_Float16)re;
    g_PsiT[(size_t)c * KDIM + DIM + s] = (_Float16)im;
  }
}

// ---------------- phase 3: WMMA GEMM C = W * Psi0R with fused |.|^2 / Z-weight epilogue ----
#define BLK 128
#define BK  32
#define LDT 40   // padded LDS stride (halves)

__global__ __launch_bounds__(256) void qnn_gemm(const float* __restrict__ dk,
                                                float* __restrict__ out){
  __shared__ __align__(16) _Float16 As[2][BLK * LDT];
  __shared__ __align__(16) _Float16 Bs[2][BLK * LDT];
  __shared__ float wsm[BLK];

  const int tid  = threadIdx.x;
  const int lane = tid & 31;
  const int wave = tid >> 5;
  const int wm   = wave >> 1;            // 0..3 : 32-row slab
  const int wn   = wave & 1;             // 0..1 : 64-col slab
  const int half = lane >> 4;            // lane group (0/1)
  const int l16  = lane & 15;
  const int m0   = blockIdx.y * BLK;     // rows of W  (2048/128 = 16)
  const int n0   = blockIdx.x * BLK;     // circuits   (8192/128 = 64)

  v8f acc[2][4];
#pragma unroll
  for (int mi = 0; mi < 2; ++mi)
#pragma unroll
    for (int ni = 0; ni < 4; ++ni)
      acc[mi][ni] = (v8f){0.f,0.f,0.f,0.f,0.f,0.f,0.f,0.f};

  // Stage one 128x32-half tile of A and B into LDS buffer `buf`.
  // 512 uint4-sized chunks per matrix; 2 per thread, coalesced along K.
  auto stage = [&](int k0, int buf){
#pragma unroll
    for (int i = 0; i < 2; ++i){
      const int u   = tid + i * 256;
      const int row = u >> 2;
      const int c4  = u & 3;
      const _Float16* gA = g_W    + (size_t)(m0 + row) * KDIM + k0 + c4 * 8;
      const _Float16* gB = g_PsiT + (size_t)(n0 + row) * KDIM + k0 + c4 * 8;
      _Float16* sA = &As[buf][row * LDT + c4 * 8];
      _Float16* sB = &Bs[buf][row * LDT + c4 * 8];
#if HAS_ASYNC_LDS
      __builtin_amdgcn_global_load_async_to_lds_b128(GPTR_V4I(gA), LPTR_V4I(sA), 0, 0);
      __builtin_amdgcn_global_load_async_to_lds_b128(GPTR_V4I(gB), LPTR_V4I(sB), 0, 0);
#else
      *(uint4*)sA = *(const uint4*)gA;
      *(uint4*)sB = *(const uint4*)gB;
#endif
    }
  };

  // Prologue: fill buffer 0.
  stage(0, 0);
#if HAS_ASYNC_LDS
  wait_async0();
#endif
  __syncthreads();

  for (int k0 = 0; k0 < KDIM; k0 += BK){
    const int buf = (k0 / BK) & 1;
    // Prefetch next tile into the other buffer while computing this one.
    if (k0 + BK < KDIM) stage(k0 + BK, buf ^ 1);

    union { v16h v; unsigned u[8]; } afrag[2], bfrag[4];
    // A 16x32 f16 layout (ISA 7.12.2): reg r<4 -> K = 2r + 8*half ; r>=4 -> K = 16 + 2(r-4) + 8*half
#pragma unroll
    for (int mi = 0; mi < 2; ++mi){
      const int row = wm * 32 + mi * 16 + l16;
#pragma unroll
      for (int r = 0; r < 8; ++r){
        const int kk = (r < 4 ? 2*r : 16 + 2*(r-4)) + 8 * half;
        afrag[mi].u[r] = *(const unsigned*)(&As[buf][row * LDT + kk]);
      }
    }
    // B 32x16 f16 layout (same K order as A): reg r<4 -> K = 2r + 8*half ; r>=4 -> K = 16 + 2(r-4) + 8*half
#pragma unroll
    for (int ni = 0; ni < 4; ++ni){
      const int row = wn * 64 + ni * 16 + l16;     // circuit within tile
#pragma unroll
      for (int r = 0; r < 8; ++r){
        const int kk = (r < 4 ? 2*r : 16 + 2*(r-4)) + 8 * half;
        bfrag[ni].u[r] = *(const unsigned*)(&Bs[buf][row * LDT + kk]);
      }
    }
#pragma unroll
    for (int mi = 0; mi < 2; ++mi)
#pragma unroll
      for (int ni = 0; ni < 4; ++ni)
        acc[mi][ni] = __builtin_amdgcn_wmma_f32_16x16x32_f16(
            false, afrag[mi].v, false, bfrag[ni].v,
            (short)0, acc[mi][ni], false, false);

#if HAS_ASYNC_LDS
    wait_async0();           // our prefetch into buf^1 has landed
#endif
    __syncthreads();         // everyone done reading buf / writing buf^1
  }

  // Epilogue: out[c] += sum_m w_{m&1023} * C[m,c]^2   (Re^2 and Im^2 rows both weighted)
  if (tid < BLK){
    const int s = (m0 + tid) & (DIM - 1);
    float w = 0.0f;
#pragma unroll
    for (int j = 0; j < NQ; ++j)
      w += ((s >> (9 - j)) & 1) ? -dk[j] : dk[j];
    wsm[tid] = w;
  }
  __syncthreads();

#pragma unroll
  for (int ni = 0; ni < 4; ++ni){
    float part = 0.0f;
#pragma unroll
    for (int mi = 0; mi < 2; ++mi){
#pragma unroll
      for (int r = 0; r < 8; ++r){
        const int mloc = wm * 32 + mi * 16 + r + 8 * half;  // C/D layout: M = r + 8*half
        const float v = acc[mi][ni][r];
        part += wsm[mloc] * v * v;
      }
    }
    // columns live in lanes L and L+16 -> butterfly, then lanes 0..15 commit
    const float other = __int_as_float(
        __builtin_amdgcn_ds_swizzle(__float_as_int(part), (0x10 << 10) | 0x1f));
    const float tot = part + other;
    if (lane < 16)
      atomicAdd(out + n0 + wn * 64 + ni * 16 + lane, tot);
  }
}

// ---------------- host launcher ----------------
extern "C" void kernel_launch(void* const* d_in, const int* in_sizes, int n_in,
                              void* d_out, int out_size, void* d_ws, size_t ws_size,
                              hipStream_t stream) {
  (void)in_sizes; (void)n_in; (void)d_ws; (void)ws_size; (void)out_size;
  const float* x  = (const float*)d_in[0];   // (8192, 10)
  const float* fm = (const float*)d_in[1];   // scalar
  const float* qw = (const float*)d_in[2];   // 69 weights
  const float* dk = (const float*)d_in[3];   // (10, 1)
  const float* db = (const float*)d_in[4];   // (1,)
  float* out = (float*)d_out;                // (8192, 1)

  qnn_init_out <<<(BATCH + 255) / 256, 256, 0, stream>>>(db, out);
  qnn_build_W  <<<DIM / 8, 256, 0, stream>>>(qw);       // 1024 waves
  qnn_build_psi<<<BATCH, 256, 0, stream>>>(x, fm);
  dim3 grid(BATCH / BLK, MDIM / BLK);                   // (64, 16)
  qnn_gemm     <<<grid, 256, 0, stream>>>(dk, out);
}
